// GATModel_55817394978867
// MI455X (gfx1250) — compile-verified, matrix operand order checked by reading, not run on hardware
//
#include <hip/hip_runtime.h>

#define DIN  256
#define DE   64
#define NH   8
#define DH   64
#define DOUT 512

typedef __attribute__((ext_vector_type(16))) __bf16 v16bf;
typedef __attribute__((ext_vector_type(8)))  float  v8f;

union V16 {        // two contiguous 16B chunks -> one WMMA bf16 operand
  v16bf bf;
  uint4 u4[2];
};

__device__ __forceinline__ unsigned short f2bf(float f) {
  unsigned u = __float_as_uint(f);
  u += 0x7FFFu + ((u >> 16) & 1u);          // round-to-nearest-even
  return (unsigned short)(u >> 16);
}

// float atomic-max via signed/unsigned integer ordering trick
__device__ __forceinline__ void atomicMaxF(float* addr, float val) {
  if (val >= 0.0f) atomicMax((int*)addr, __float_as_int(val));
  else             atomicMin((unsigned int*)addr, __float_as_uint(val));
}

// ---------------------------------------------------------------------------
// Pre-pass: fp32 -> bf16 elementwise (n multiple of 4)
// ---------------------------------------------------------------------------
__global__ __launch_bounds__(256)
void cvt_bf16(const float* __restrict__ in, unsigned short* __restrict__ outp, long n) {
  long base = ((long)blockIdx.x * 256 + threadIdx.x) * 4;
  if (base + 3 < n) {
    float4 f = *(const float4*)(in + base);
    ushort4 o;
    o.x = f2bf(f.x); o.y = f2bf(f.y); o.z = f2bf(f.z); o.w = f2bf(f.w);
    *(ushort4*)(outp + base) = o;
  }
}

// Pre-pass: W[K x Nc] fp32 -> WT[Nc x K] bf16 (tiny, one-shot)
__global__ __launch_bounds__(256)
void transpose_cvt_bf16(const float* __restrict__ W, unsigned short* __restrict__ WT,
                        int K, int Nc) {
  int i = blockIdx.x * 256 + threadIdx.x;
  if (i >= K * Nc) return;
  int n = i / K, kk = i - n * K;
  WT[i] = f2bf(W[(size_t)kk * Nc + n]);
}

// ---------------------------------------------------------------------------
// C[M x Nc] = A[M x K] * B[K x Nc]; A bf16 row-major, BT bf16 [Nc x K].
// One wave -> 16 x 64 output tile (A loads amortized over 4 WMMA col tiles).
// Lane l16 of half `half` holds K elems {koff..+7, koff+16..+23} -> 2x b128.
// ---------------------------------------------------------------------------
__global__ __launch_bounds__(32)
void gemm_bf16_wmma(const unsigned short* __restrict__ A,
                    const unsigned short* __restrict__ BT,
                    float* __restrict__ C, int K, int Nc) {
  const int tm   = blockIdx.x;
  const int n0   = blockIdx.y * 64;
  const int lane = threadIdx.x & 31;
  const int half = lane >> 4;
  const int l16  = lane & 15;
  const int koff = half * 8;

  v8f acc[4] = {};
  const unsigned short* arow = A + (size_t)(tm * 16 + l16) * K + koff;
  for (int k0 = 0; k0 < K; k0 += 32) {
    V16 a;
    a.u4[0] = *(const uint4*)(arow + k0);
    a.u4[1] = *(const uint4*)(arow + k0 + 16);
#pragma unroll
    for (int ct = 0; ct < 4; ++ct) {
      const unsigned short* brow = BT + (size_t)(n0 + ct * 16 + l16) * K + k0 + koff;
      V16 b;
      b.u4[0] = *(const uint4*)(brow);
      b.u4[1] = *(const uint4*)(brow + 16);
      acc[ct] = __builtin_amdgcn_wmma_f32_16x16x32_bf16(false, a.bf, false, b.bf,
                                                        (short)0, acc[ct], false, false);
    }
  }
#pragma unroll
  for (int ct = 0; ct < 4; ++ct)
#pragma unroll
    for (int j = 0; j < 8; ++j)
      C[(size_t)(tm * 16 + j + 8 * half) * Nc + n0 + ct * 16 + l16] = acc[ct][j];
}

// ---------------------------------------------------------------------------
// Fused edge stage: per (16-edge tile, head): ke = EF@We[:,h] via WMMA in
// registers, gather k[src]/q[dst] into the same C layout, dot-reduce to
// logits[e,h], atomic-max into m[dst,h]. Avoids 819 MB ke intermediate.
// ---------------------------------------------------------------------------
__global__ __launch_bounds__(32)
void edge_logits_wmma(const unsigned short* __restrict__ EF,   // [E x 64] bf16
                      const unsigned short* __restrict__ WeT,  // [512 x 64] bf16
                      const float* __restrict__ Q,  const float* __restrict__ Km,
                      const int* __restrict__ src,  const int* __restrict__ dst,
                      float* __restrict__ logits,   float* __restrict__ mmax) {
  const int et   = blockIdx.x;
  const int h    = blockIdx.y;
  const int e0   = et * 16;
  const int lane = threadIdx.x & 31;
  const int half = lane >> 4;
  const int l16  = lane & 15;
  const int koff = half * 8;

  v8f acc[4] = {};
  const unsigned short* arow = EF + (size_t)(e0 + l16) * DE + koff;
#pragma unroll
  for (int k0 = 0; k0 < DE; k0 += 32) {
    V16 a;
    a.u4[0] = *(const uint4*)(arow + k0);
    a.u4[1] = *(const uint4*)(arow + k0 + 16);
#pragma unroll
    for (int ct = 0; ct < 4; ++ct) {
      const unsigned short* brow =
          WeT + (size_t)(h * DH + ct * 16 + l16) * DE + k0 + koff;
      V16 b;
      b.u4[0] = *(const uint4*)(brow);
      b.u4[1] = *(const uint4*)(brow + 16);
      acc[ct] = __builtin_amdgcn_wmma_f32_16x16x32_bf16(false, a.bf, false, b.bf,
                                                        (short)0, acc[ct], false, false);
    }
  }

  // rows this lane holds in C layout: j + 8*half
  int sidx[8], didx[8];
#pragma unroll
  for (int j = 0; j < 8; ++j) {
    sidx[j] = src[e0 + j + 8 * half];
    didx[j] = dst[e0 + j + 8 * half];
  }

  float part[8] = {0.f, 0.f, 0.f, 0.f, 0.f, 0.f, 0.f, 0.f};
#pragma unroll
  for (int ct = 0; ct < 4; ++ct) {
    const int coln = h * DH + ct * 16 + l16;
#pragma unroll
    for (int j = 0; j < 8; ++j) {
      float kg = Km[(size_t)sidx[j] * DOUT + coln];
      float qg = Q [(size_t)didx[j] * DOUT + coln];
      part[j] += qg * (acc[ct][j] + kg);
    }
  }
  // reduce across the 16-lane group (xor masks < 16 stay within each half)
#pragma unroll
  for (int j = 0; j < 8; ++j) {
    part[j] += __shfl_xor(part[j], 1, 32);
    part[j] += __shfl_xor(part[j], 2, 32);
    part[j] += __shfl_xor(part[j], 4, 32);
    part[j] += __shfl_xor(part[j], 8, 32);
  }
  if (l16 == 0) {
    const float scale = 0.125f;   // 1/sqrt(DH)
#pragma unroll
    for (int j = 0; j < 8; ++j) {
      const int row = e0 + j + 8 * half;
      const float lg = part[j] * scale;
      logits[(size_t)row * NH + h] = lg;
      atomicMaxF(&mmax[(size_t)didx[j] * NH + h], lg);
    }
  }
}

// m = -inf (bit pattern works for the int-ordering atomic-max trick)
__global__ void init_m(unsigned* __restrict__ m, int n) {
  int i = blockIdx.x * blockDim.x + threadIdx.x;
  if (i < n) m[i] = 0xFF800000u;
}

// after logits: zero agg (aliases q) + denom, replace non-finite m with 0
__global__ void init_post(float* __restrict__ agg, float* __restrict__ denom,
                          float* __restrict__ m, int nAgg, int nH) {
  int i = blockIdx.x * blockDim.x + threadIdx.x;
  if (i < nAgg) agg[i] = 0.0f;
  if (i < nH) {
    denom[i] = 0.0f;
    float mv = m[i];
    bool finite = (__float_as_uint(mv) & 0x7F800000u) != 0x7F800000u;
    m[i] = finite ? mv : 0.0f;
  }
}

// w = exp(logit - m[dst]); denom[dst,h] += w; agg[dst,:] += w * v[src,:]
__global__ __launch_bounds__(256)
void scatter_agg(const float* __restrict__ V, const int* __restrict__ src,
                 const int* __restrict__ dst, const float* __restrict__ logits,
                 const float* __restrict__ m, float* __restrict__ agg,
                 float* __restrict__ denom, int E_) {
  int gid = blockIdx.x * 256 + threadIdx.x;
  int e = gid >> 6;           // 64 threads per edge
  int t = gid & 63;
  if (e >= E_) return;
  int s = src[e], d = dst[e];
  int h = t >> 3;
  float w = __expf(logits[(size_t)e * NH + h] - m[(size_t)d * NH + h]);
  int cbase = t * 8;
  const float4* vp = (const float4*)(V + (size_t)s * DOUT + cbase);
  float4 v0 = vp[0], v1 = vp[1];
  float* ap = agg + (size_t)d * DOUT + cbase;
  atomicAdd(ap + 0, w * v0.x); atomicAdd(ap + 1, w * v0.y);
  atomicAdd(ap + 2, w * v0.z); atomicAdd(ap + 3, w * v0.w);
  atomicAdd(ap + 4, w * v1.x); atomicAdd(ap + 5, w * v1.y);
  atomicAdd(ap + 6, w * v1.z); atomicAdd(ap + 7, w * v1.w);
  if (t < 8) {
    float w2 = __expf(logits[(size_t)e * NH + t] - m[(size_t)d * NH + t]);
    atomicAdd(denom + (size_t)d * NH + t, w2);
  }
}

// out[n] = sigmoid( relu(agg/denom) . Wd + bd ), one wave per node
__global__ __launch_bounds__(256)
void final_out(const float* __restrict__ agg, const float* __restrict__ denom,
               const float* __restrict__ Wd, const float* __restrict__ bd,
               float* __restrict__ out, int Nn) {
  int wave = (blockIdx.x * 256 + threadIdx.x) >> 5;
  int lane = threadIdx.x & 31;
  if (wave >= Nn) return;
  const float* a  = agg   + (size_t)wave * DOUT;
  const float* dn = denom + (size_t)wave * NH;
  float s = 0.0f;
#pragma unroll
  for (int i = 0; i < 16; ++i) {
    int c = lane + i * 32;           // coalesced
    int h = c >> 6;
    float x = a[c] / (dn[h] + 1e-9f);
    x = fmaxf(x, 0.0f);
    s += x * Wd[c];
  }
  s += __shfl_xor(s, 16, 32);
  s += __shfl_xor(s, 8, 32);
  s += __shfl_xor(s, 4, 32);
  s += __shfl_xor(s, 2, 32);
  s += __shfl_xor(s, 1, 32);
  if (lane == 0) out[wave] = 1.0f / (1.0f + __expf(-(s + bd[0])));
}

extern "C" void kernel_launch(void* const* d_in, const int* in_sizes, int n_in,
                              void* d_out, int out_size, void* d_ws, size_t ws_size,
                              hipStream_t stream) {
  const float* nf = (const float*)d_in[0];
  const float* ef = (const float*)d_in[1];
  const float* Wq = (const float*)d_in[2];
  const float* Wk = (const float*)d_in[3];
  const float* Wv = (const float*)d_in[4];
  const float* We = (const float*)d_in[5];
  const float* Wd = (const float*)d_in[6];
  const float* bd = (const float*)d_in[7];
  const int*   src = (const int*)d_in[8];
  const int*   dst = (const int*)d_in[9];
  float* out = (float*)d_out;

  const int N = in_sizes[0] / DIN;   // 50000
  const int E = in_sizes[8];         // 400000

  // ---- workspace layout (agg aliases q: q is dead after edge_logits) ----
  char* ws = (char*)d_ws;
  size_t off = 0;
  const size_t SZ_QKV = (size_t)N * DOUT * sizeof(float);   // 102.4 MB each
  float* q      = (float*)(ws + off); off += SZ_QKV;
  float* k      = (float*)(ws + off); off += SZ_QKV;
  float* v      = (float*)(ws + off); off += SZ_QKV;
  float* logits = (float*)(ws + off); off += (size_t)E * NH * sizeof(float);
  float* m      = (float*)(ws + off); off += (size_t)N * NH * sizeof(float);
  float* denom  = (float*)(ws + off); off += (size_t)N * NH * sizeof(float);
  unsigned short* nf_bf  = (unsigned short*)(ws + off); off += (size_t)N * DIN * 2;
  unsigned short* ef_bf  = (unsigned short*)(ws + off); off += (size_t)E * DE * 2;
  unsigned short* WqT_bf = (unsigned short*)(ws + off); off += (size_t)DOUT * DIN * 2;
  unsigned short* WkT_bf = (unsigned short*)(ws + off); off += (size_t)DOUT * DIN * 2;
  unsigned short* WvT_bf = (unsigned short*)(ws + off); off += (size_t)DOUT * DIN * 2;
  unsigned short* WeT_bf = (unsigned short*)(ws + off); off += (size_t)DOUT * DE * 2;
  float* agg = q;   // alias

  // ---- pre-passes: convert activations, transpose+convert weights ----
  long nNF = (long)N * DIN, nEF = (long)E * DE;
  cvt_bf16<<<(unsigned)((nNF / 4 + 255) / 256), 256, 0, stream>>>(nf, nf_bf, nNF);
  cvt_bf16<<<(unsigned)((nEF / 4 + 255) / 256), 256, 0, stream>>>(ef, ef_bf, nEF);
  int wqn = DIN * DOUT, wen = DE * DOUT;
  transpose_cvt_bf16<<<(wqn + 255) / 256, 256, 0, stream>>>(Wq, WqT_bf, DIN, DOUT);
  transpose_cvt_bf16<<<(wqn + 255) / 256, 256, 0, stream>>>(Wk, WkT_bf, DIN, DOUT);
  transpose_cvt_bf16<<<(wqn + 255) / 256, 256, 0, stream>>>(Wv, WvT_bf, DIN, DOUT);
  transpose_cvt_bf16<<<(wen + 255) / 256, 256, 0, stream>>>(We, WeT_bf, DE, DOUT);

  // ---- q/k/v GEMMs (bf16 WMMA, fp32 accumulate) ----
  dim3 gGemm(N / 16, DOUT / 64);     // 3125 x 8
  gemm_bf16_wmma<<<gGemm, 32, 0, stream>>>(nf_bf, WqT_bf, q, DIN, DOUT);
  gemm_bf16_wmma<<<gGemm, 32, 0, stream>>>(nf_bf, WkT_bf, k, DIN, DOUT);
  gemm_bf16_wmma<<<gGemm, 32, 0, stream>>>(nf_bf, WvT_bf, v, DIN, DOUT);

  init_m<<<(N * NH + 255) / 256, 256, 0, stream>>>((unsigned*)m, N * NH);

  // ---- fused edge-key GEMM + attention logits + segment max ----
  dim3 gEdge(E / 16, NH);            // 25000 x 8
  edge_logits_wmma<<<gEdge, 32, 0, stream>>>(ef_bf, WeT_bf, q, k, src, dst, logits, m);

  init_post<<<((size_t)N * DOUT + 255) / 256, 256, 0, stream>>>(
      agg, denom, m, N * DOUT, N * NH);

  // ---- softmax weights + atomic scatter aggregation ----
  scatter_agg<<<((size_t)E * 64 + 255) / 256, 256, 0, stream>>>(
      v, src, dst, logits, m, agg, denom, E);

  // ---- relu + Dense(1) + sigmoid ----
  final_out<<<(N * 32 + 255) / 256, 256, 0, stream>>>(agg, denom, Wd, bd, out, N);
}
